// VKDE_18476949307509
// MI455X (gfx1250) — matrix operand order repaired and checked by reading, not compiled, not run on hardware
//
#include <hip/hip_runtime.h>
#include <math.h>

// ---------------- problem dims ----------------
#define BN   128
#define KN   32
#define IN   12000
#define HN   600
#define LATN 200
#define NOUT2 400              // 2*LAT
#define MROWS 4096             // B*K
// padded (multiple-of-32 K dims for WMMA, 16-wide N tiles)
#define HP      608            // 600 -> 19*32
#define LATP    224            // 200 -> 7*32
#define W1TROWS 640            // 600 -> 10 x 64-wide N chunks
#define ITNROWS 12032          // 12000 -> 188 x 64-wide N chunks
#define KSTEPS  (IN / 32)      // 375

typedef __attribute__((ext_vector_type(16))) __bf16         v16bf;
typedef __attribute__((ext_vector_type(8)))  float          v8f;
typedef __attribute__((ext_vector_type(8)))  unsigned short v8u;
typedef __attribute__((ext_vector_type(16))) unsigned short v16u;
typedef __attribute__((ext_vector_type(4)))  unsigned int   v4ui;
typedef __attribute__((ext_vector_type(8)))  int            v8si;
typedef __attribute__((ext_vector_type(4)))  int            v4si;

// ---------------- helpers ----------------
__device__ __forceinline__ unsigned short f2bf(float f) {
  unsigned int u = __float_as_uint(f);
  u += 0x7FFFu + ((u >> 16) & 1u);          // round-to-nearest-even
  return (unsigned short)(u >> 16);
}

// build a 16-element bf16 fragment from two 16-byte (8 x u16) aligned loads
__device__ __forceinline__ v16bf mkfrag(const unsigned short* lo,
                                        const unsigned short* hi) {
  v8u a = *(const v8u*)lo;
  v8u b = *(const v8u*)hi;
  v16u t;
#pragma unroll
  for (int i = 0; i < 8; ++i) { t[i] = a[i]; t[i + 8] = b[i]; }
  return __builtin_bit_cast(v16bf, t);
}

__device__ __forceinline__ v8f wmma_bf16(v16bf a, v16bf b, v8f c) {
  return __builtin_amdgcn_wmma_f32_16x16x32_bf16(false, a, false, b,
                                                 (short)0, c, false, false);
}

// ---------------- workspace layout (bytes) ----------------
constexpr size_t alignup(size_t x) { return (x + 255) & ~size_t(255); }
constexpr size_t XH_OFF  = 0;                                                   // bf16 x [4096][12000]
constexpr size_t W1T_OFF = alignup(XH_OFF  + (size_t)MROWS * IN * 2);           // bf16 W1^T [640][12000]
constexpr size_t H_OFF   = alignup(W1T_OFF + (size_t)W1TROWS * IN * 2);         // bf16 h [4096][608]
constexpr size_t W2T_OFF = alignup(H_OFF   + (size_t)MROWS * HP * 2);           // bf16 W2^T [400][608]
constexpr size_t ENC_OFF = alignup(W2T_OFF + (size_t)NOUT2 * HP * 2);           // f32 enc [4096][400]
constexpr size_t ZN_OFF  = alignup(ENC_OFF + (size_t)MROWS * NOUT2 * 4);        // bf16 z_n [4096][224]
constexpr size_t ITN_OFF = alignup(ZN_OFF  + (size_t)MROWS * LATP * 2);         // bf16 items_n [12032][224]
constexpr size_t KLP_OFF = alignup(ITN_OFF + (size_t)ITNROWS * LATP * 2);       // f32 klpart [4096]

// out element offsets (tuple concat: z, new_output, kl, x)
constexpr long long Z_OFF    = 0;
constexpr long long NOUT_OFF = (long long)MROWS * LATN;                  // 819200
constexpr long long KL_OFF   = NOUT_OFF + (long long)BN * IN;            // 2355200
constexpr long long X_OFF    = KL_OFF + 1;                               // 2355201

// ---------------- prep kernels ----------------
__global__ void prep_w1t(const float* __restrict__ W1, unsigned short* __restrict__ w1t) {
  long long i = (long long)blockIdx.x * 256 + threadIdx.x;
  if (i >= (long long)W1TROWS * IN) return;
  int n = (int)(i / IN), k = (int)(i % IN);
  float v = (n < HN) ? W1[(long long)k * HN + n] : 0.f;
  w1t[i] = f2bf(v);
}

__global__ void prep_w2t(const float* __restrict__ W2, unsigned short* __restrict__ w2t) {
  int i = blockIdx.x * 256 + threadIdx.x;
  if (i >= NOUT2 * HP) return;
  int n = i / HP, k = i % HP;
  float v = (k < HN) ? W2[(long long)k * NOUT2 + n] : 0.f;
  w2t[i] = f2bf(v);
}

// l2-normalize item rows -> bf16 [12032][224] (zero-padded K and rows)
__global__ void prep_items(const float* __restrict__ items, unsigned short* __restrict__ itn) {
  int wave = threadIdx.x >> 5, lane = threadIdx.x & 31;
  int row = blockIdx.x * 8 + wave;              // 1504 blocks * 8 waves = 12032
  const float* p = items + (long long)row * LATN;
  float v[7]; float ss = 0.f;
#pragma unroll
  for (int t = 0; t < 7; ++t) {
    int j = lane + t * 32;
    v[t] = (row < IN && j < LATN) ? p[j] : 0.f;
    ss += v[t] * v[t];
  }
#pragma unroll
  for (int s = 16; s > 0; s >>= 1) ss += __shfl_xor(ss, s, 32);
  float inv = 1.f / fmaxf(sqrtf(ss), 1e-12f);
#pragma unroll
  for (int t = 0; t < 7; ++t) {
    int j = lane + t * 32;                       // covers 0..223
    itn[(long long)row * LATP + j] = f2bf(v[t] * inv);
  }
}

// ---------------- x = l2(l1(gram[idx] * (rating>0))) ----------------
__global__ void x_kernel(const float* __restrict__ rating, const int* __restrict__ items_idx,
                         const float* __restrict__ gram,
                         float* __restrict__ xout, unsigned short* __restrict__ xh) {
  int row = blockIdx.x;                 // 0..4095
  int b = row >> 5;
  int idx = items_idx[row];
  const float* g = gram + (long long)idx * IN;
  const float* r = rating + (long long)b * IN;
  float s1 = 0.f, s2 = 0.f;
  for (int i = threadIdx.x; i < IN; i += 256) {
    float v = g[i];
    v = (r[i] > 0.f) ? v : 0.f;
    s1 += fabsf(v);
    s2 += v * v;
  }
  __shared__ float sh1[256], sh2[256];
  sh1[threadIdx.x] = s1; sh2[threadIdx.x] = s2; __syncthreads();
  for (int s = 128; s > 0; s >>= 1) {
    if (threadIdx.x < (unsigned)s) {
      sh1[threadIdx.x] += sh1[threadIdx.x + s];
      sh2[threadIdx.x] += sh2[threadIdx.x + s];
    }
    __syncthreads();
  }
  float S1 = fmaxf(sh1[0], 1e-12f);
  float l2 = sqrtf(sh2[0]) / S1;
  float scale = 1.f / (S1 * fmaxf(l2, 1e-12f));
  for (int i = threadIdx.x; i < IN; i += 256) {
    float v = g[i];
    v = (r[i] > 0.f) ? v : 0.f;
    v *= scale;
    xout[(long long)row * IN + i] = v;            // f32 output (odd base offset -> scalar stores)
    xh[(long long)row * IN + i] = f2bf(v);        // bf16 copy for WMMA
  }
}

// ---------------- GEMM1: h = tanh(x @ W1 + b1) ----------------
// block = 4 waves = 64M x 64N; wave tile 16M x 64N (4 accumulators share A).
// Shared 64Nx32K bf16 B-tile staged per K-step by the Tensor Data Mover
// (tensor_load_to_lds, double-buffered, s_wait_tensorcnt + barrier).
__global__ void __launch_bounds__(128) gemm1_kernel(const unsigned short* __restrict__ xh,
                             const unsigned short* __restrict__ w1t,
                             const float* __restrict__ b1,
                             unsigned short* __restrict__ h) {
  __shared__ unsigned short bstage[2][64 * 32];   // [buf][n][k], 8 KB total
  int wave = threadIdx.x >> 5, lane = threadIdx.x & 31;
  int half = lane >> 4, l15 = lane & 15;
  int m0 = blockIdx.x * 64 + wave * 16;
  int nbase = blockIdx.y * 64;                    // < 640

  const unsigned short* ap = xh + (long long)(m0 + l15) * IN + half * 16;

  // --- TDM descriptor issue (2D: tile 32K x 64N, row stride = IN elements) ---
  auto tdm_issue = [&](int buf, int kk) {
    unsigned long long ga =
        (unsigned long long)(const unsigned short*)(w1t + (long long)nbase * IN + kk);
    unsigned lds = (unsigned)(unsigned long long)&bstage[buf][0];
    v4ui g0;
    g0[0] = 1u;                                        // count=1, user descriptor
    g0[1] = lds;                                       // lds_addr
    g0[2] = (unsigned)(ga & 0xFFFFFFFFu);              // global_addr[31:0]
    g0[3] = (unsigned)((ga >> 32) & 0x01FFFFFFu)       // global_addr[56:32]
          | (2u << 30);                                // type=2 ("image")
    v8si g1;
    g1[0] = (int)(1u << 16);       // workgroup_mask=0, data_size=1 (2 bytes)
    g1[1] = (int)(32u << 16);      // tensor_dim0 = 32 (bits 79:48, low half)
    g1[2] = (int)(64u << 16);      // tensor_dim0 hi=0 | tensor_dim1 = 64 (low half)
    g1[3] = (int)(32u << 16);      // tensor_dim1 hi=0 | tile_dim0 = 32
    g1[4] = 64;                    // tile_dim1 = 64, tile_dim2 = 0
    g1[5] = IN;                    // tensor_dim0_stride[31:0] = 12000
    g1[6] = 0;                     // stride hi / tensor_dim1_stride lo = 0
    g1[7] = 0;
    v4si g2 = {0, 0, 0, 0}, g3 = {0, 0, 0, 0};
    v8si g4 = {0, 0, 0, 0, 0, 0, 0, 0};
    __builtin_amdgcn_tensor_load_to_lds(g0, g1, g2, g3, g4, 0);   // 6-arg toolchain form
  };

  if (wave == 0) tdm_issue(0, 0);

  v8f c0 = {}, c1 = {}, c2 = {}, c3 = {};
  for (int i = 0; i < KSTEPS; ++i) {
    if (wave == 0) __builtin_amdgcn_s_wait_tensorcnt((short)0);  // buf[i&1] ready
    __syncthreads();                                             // publish to all waves
    if (wave == 0 && i + 1 < KSTEPS) tdm_issue((i + 1) & 1, (i + 1) * 32);

    v16bf a = mkfrag(ap, ap + 8);
    ap += 32;
    __builtin_prefetch(ap + 64, 0, 1);             // global_prefetch_b8 for A stream

    const unsigned short* lb = &bstage[i & 1][0];
    {
      const unsigned short* bp = lb + (0 * 16 + l15) * 32 + half * 16;
      c0 = wmma_bf16(a, mkfrag(bp, bp + 8), c0);
    }
    {
      const unsigned short* bp = lb + (1 * 16 + l15) * 32 + half * 16;
      c1 = wmma_bf16(a, mkfrag(bp, bp + 8), c1);
    }
    {
      const unsigned short* bp = lb + (2 * 16 + l15) * 32 + half * 16;
      c2 = wmma_bf16(a, mkfrag(bp, bp + 8), c2);
    }
    {
      const unsigned short* bp = lb + (3 * 16 + l15) * 32 + half * 16;
      c3 = wmma_bf16(a, mkfrag(bp, bp + 8), c3);
    }
    // next TDM write to buf[i&1] is issued after the top barrier of iter i+1,
    // which orders it after all reads above -> no trailing barrier needed
  }

  v8f acc[4] = {c0, c1, c2, c3};
#pragma unroll
  for (int g = 0; g < 4; ++g) {
    int n = nbase + g * 16 + l15;
    float bias = (n < HN) ? b1[n] : 0.f;
#pragma unroll
    for (int rr = 0; rr < 8; ++rr) {
      int mr = m0 + rr + 8 * half;                 // C layout: VGPR rr -> M=rr(+8), N=l15
      if (n < HN)      h[(long long)mr * HP + n] = f2bf(tanhf(acc[g][rr] + bias));
      else if (n < HP) h[(long long)mr * HP + n] = 0;   // zero K-pad for GEMM2
    }
  }
}

// ---------------- GEMM2: enc = h @ W2 + b2 (f32 out) ----------------
__global__ void gemm2_kernel(const unsigned short* __restrict__ h,
                             const unsigned short* __restrict__ w2t,
                             const float* __restrict__ b2,
                             float* __restrict__ enc) {
  int wave = threadIdx.x >> 5, lane = threadIdx.x & 31;
  int half = lane >> 4, l15 = lane & 15;
  int tile = blockIdx.y * 4 + wave;
  if (tile >= NOUT2 / 16) return;                // 25 tiles, wave-uniform guard
  int m0 = blockIdx.x * 16, n0 = tile * 16;
  const unsigned short* ap = h   + (long long)(m0 + l15) * HP + half * 16;
  const unsigned short* bp = w2t + (long long)(n0 + l15) * HP + half * 16;
  v8f c = {};
  for (int kk = 0; kk < HP; kk += 32) {
    v16bf a  = mkfrag(ap, ap + 8);
    v16bf bb = mkfrag(bp, bp + 8);
    c = wmma_bf16(a, bb, c);
    ap += 32; bp += 32;
  }
  int n = n0 + l15;
  float bias = b2[n];
#pragma unroll
  for (int rr = 0; rr < 8; ++rr) {
    int mr = m0 + rr + 8 * half;
    enc[(long long)mr * NOUT2 + n] = c[rr] + bias;
  }
}

// ---------------- z / z_n / kl partials ----------------
__global__ void z_kernel(const float* __restrict__ enc, float* __restrict__ zout,
                         unsigned short* __restrict__ zn, float* __restrict__ klpart) {
  int m = blockIdx.x, tid = threadIdx.x;
  const float* row = enc + (long long)m * NOUT2;
  float v0 = row[tid];                           // tid < 256 < 400: always valid
  int j1 = tid + 256;
  float v1 = (j1 < NOUT2) ? row[j1] : 0.f;
  float nrm = 0.f, kl = 0.f;
  if (tid < LATN) { nrm = v0 * v0; kl = v0 * v0; }           // mean cols
  else            { kl = expf(v0) - 1.f - v0; }              // logvar cols 200..255
  if (j1 < NOUT2) kl += expf(v1) - 1.f - v1;                 // logvar cols 256..399
  __shared__ float shn[256], shk[256];
  shn[tid] = nrm; shk[tid] = kl; __syncthreads();
  for (int s = 128; s > 0; s >>= 1) {
    if (tid < s) { shn[tid] += shn[tid + s]; shk[tid] += shk[tid + s]; }
    __syncthreads();
  }
  float inv = 1.f / fmaxf(sqrtf(shn[0]), 1e-12f);
  if (tid < LATN) {
    zout[(long long)m * LATN + tid] = v0;                    // z = mean (f32 output)
    zn[(long long)m * LATP + tid] = f2bf(v0 * inv);          // normalized bf16
  } else if (tid < LATP) {
    zn[(long long)m * LATP + tid] = 0;                       // zero K-pad
  }
  if (tid == 0) klpart[m] = shk[0];
}

__global__ void klfinal_kernel(const float* __restrict__ klpart, float* __restrict__ out_kl) {
  __shared__ float sh[256];
  float s = 0.f;
  for (int i = threadIdx.x; i < MROWS; i += 256) s += klpart[i];   // fixed-order: deterministic
  sh[threadIdx.x] = s; __syncthreads();
  for (int st = 128; st > 0; st >>= 1) {
    if (threadIdx.x < (unsigned)st) sh[threadIdx.x] += sh[threadIdx.x + st];
    __syncthreads();
  }
  if (threadIdx.x == 0) out_kl[0] = 0.5f * sh[0] / (float)MROWS;
}

// ---------------- GEMM3 fused: new_output = log(mean_K exp(z_n@items_n^T/tau)+1) ----------------
// wave tile 32(M) x 64(N): 8 accumulators; each wave owns full K=32 user segments
// for its 64 columns -> deterministic fused reduce, no atomics.
__global__ void __launch_bounds__(128) gemm3_kernel(const unsigned short* __restrict__ zn,
                             const unsigned short* __restrict__ itn,
                             float* __restrict__ nout) {
  int wave = threadIdx.x >> 5, lane = threadIdx.x & 31;
  int half = lane >> 4, l15 = lane & 15;
  int m0 = blockIdx.x * 32;
  int n0 = (blockIdx.y * 4 + wave) * 64;         // up to 12032 - 64 (rows padded)
  const unsigned short* ap0 = zn  + (long long)(m0 + l15) * LATP + half * 16;
  const unsigned short* ap1 = ap0 + 16 * LATP;
  const unsigned short* bp0 = itn + (long long)(n0 +  0 + l15) * LATP + half * 16;
  const unsigned short* bp1 = itn + (long long)(n0 + 16 + l15) * LATP + half * 16;
  const unsigned short* bp2 = itn + (long long)(n0 + 32 + l15) * LATP + half * 16;
  const unsigned short* bp3 = itn + (long long)(n0 + 48 + l15) * LATP + half * 16;
  v8f c00 = {}, c01 = {}, c10 = {}, c11 = {}, c20 = {}, c21 = {}, c30 = {}, c31 = {};
  for (int kk = 0; kk < LATP; kk += 32) {
    v16bf a0 = mkfrag(ap0, ap0 + 8);
    v16bf a1 = mkfrag(ap1, ap1 + 8);
    v16bf b0 = mkfrag(bp0, bp0 + 8);
    c00 = wmma_bf16(a0, b0, c00);  c01 = wmma_bf16(a1, b0, c01);
    v16bf b1 = mkfrag(bp1, bp1 + 8);
    c10 = wmma_bf16(a0, b1, c10);  c11 = wmma_bf16(a1, b1, c11);
    v16bf b2 = mkfrag(bp2, bp2 + 8);
    c20 = wmma_bf16(a0, b2, c20);  c21 = wmma_bf16(a1, b2, c21);
    v16bf b3 = mkfrag(bp3, bp3 + 8);
    c30 = wmma_bf16(a0, b3, c30);  c31 = wmma_bf16(a1, b3, c31);
    ap0 += 32; ap1 += 32; bp0 += 32; bp1 += 32; bp2 += 32; bp3 += 32;
  }
  int b = m0 >> 5;
  v8f g0[4] = {c00, c10, c20, c30};
  v8f g1[4] = {c01, c11, c21, c31};
#pragma unroll
  for (int g = 0; g < 4; ++g) {
    float s0 = 0.f, s1 = 0.f;
#pragma unroll
    for (int rr = 0; rr < 8; ++rr) {
      s0 += expf(g0[g][rr] * 10.f);              // 1/TAU = 10
      s1 += expf(g1[g][rr] * 10.f);
    }
    s0 += __shfl_xor(s0, 16, 32);                // combine M halves of each C tile
    s1 += __shfl_xor(s1, 16, 32);
    float tot = s0 + s1;                         // sum over 32 rows (one user)
    int col = n0 + g * 16 + l15;
    if (lane < 16 && col < IN) {
      nout[(long long)b * IN + col] = logf(tot * (1.f / 32.f) + 1.f);
    }
  }
}

// ---------------- launch ----------------
extern "C" void kernel_launch(void* const* d_in, const int* in_sizes, int n_in,
                              void* d_out, int out_size, void* d_ws, size_t ws_size,
                              hipStream_t stream) {
  (void)in_sizes; (void)n_in; (void)out_size; (void)ws_size;
  const float* rating   = (const float*)d_in[0];
  const int*   items_ix = (const int*)d_in[1];
  const float* gram     = (const float*)d_in[2];
  const float* W1       = (const float*)d_in[3];
  const float* b1       = (const float*)d_in[4];
  const float* W2       = (const float*)d_in[5];
  const float* b2       = (const float*)d_in[6];
  const float* items    = (const float*)d_in[7];

  char* ws = (char*)d_ws;
  unsigned short* xh  = (unsigned short*)(ws + XH_OFF);
  unsigned short* w1t = (unsigned short*)(ws + W1T_OFF);
  unsigned short* hbuf= (unsigned short*)(ws + H_OFF);
  unsigned short* w2t = (unsigned short*)(ws + W2T_OFF);
  float*          enc = (float*)(ws + ENC_OFF);
  unsigned short* zn  = (unsigned short*)(ws + ZN_OFF);
  unsigned short* itn = (unsigned short*)(ws + ITN_OFF);
  float*          klp = (float*)(ws + KLP_OFF);

  float* out    = (float*)d_out;
  float* z_out  = out + Z_OFF;
  float* n_out  = out + NOUT_OFF;
  float* kl_out = out + KL_OFF;
  float* x_out  = out + X_OFF;

  prep_w1t<<<(W1TROWS * IN + 255) / 256, 256, 0, stream>>>(W1, w1t);
  prep_w2t<<<(NOUT2 * HP + 255) / 256, 256, 0, stream>>>(W2, w2t);
  prep_items<<<ITNROWS / 8, 256, 0, stream>>>(items, itn);
  x_kernel<<<MROWS, 256, 0, stream>>>(rating, items_ix, gram, x_out, xh);
  gemm1_kernel<<<dim3(MROWS / 64, W1TROWS / 64), 128, 0, stream>>>(xh, w1t, b1, hbuf);
  gemm2_kernel<<<dim3(MROWS / 16, 7), 128, 0, stream>>>(hbuf, w2t, b2, enc);
  z_kernel<<<MROWS, 256, 0, stream>>>(enc, z_out, zn, klp);
  klfinal_kernel<<<1, 256, 0, stream>>>(klp, kl_out);
  gemm3_kernel<<<dim3(MROWS / 32, 47), 128, 0, stream>>>(zn, itn, n_out);
}
